// ResNet_10333691314349
// MI455X (gfx1250) — matrix operand order repaired and checked by reading, not compiled, not run on hardware
//
#include <hip/hip_runtime.h>
#include <hip/hip_bf16.h>

typedef __attribute__((ext_vector_type(16))) _Float16 v16h;
typedef __attribute__((ext_vector_type(8)))  _Float16 v8h;
typedef __attribute__((ext_vector_type(8)))  float    v8f;

#define BATCH 128
#define SEQ   96
#define DIM   400

// ---------------------------------------------------------------------------
// Embedding gather: xc[b][n][0:300]=emb[x-1] (0 if x==0),
// [300:350]=pos[dist1] (0 if ==191), [350:400]=pos[dist2]. Written as f16.
// ---------------------------------------------------------------------------
__global__ void gather_embed_kernel(const int* __restrict__ x,
                                    const int* __restrict__ d1,
                                    const int* __restrict__ d2,
                                    const float* __restrict__ emb,
                                    const float* __restrict__ pos,
                                    _Float16* __restrict__ xc, int total)
{
    int idx = blockIdx.x * blockDim.x + threadIdx.x;
    if (idx >= total) return;
    int d  = idx % DIM;
    int bn = idx / DIM;
    float v;
    if (d < 300) {
        int xi = x[bn];
        v = (xi == 0) ? 0.f : emb[(size_t)(xi - 1) * 300 + d];
    } else if (d < 350) {
        int di = d1[bn];
        v = (di == 191) ? 0.f : pos[di * 50 + (d - 300)];
    } else {
        int di = d2[bn];
        v = (di == 191) ? 0.f : pos[di * 50 + (d - 350)];
    }
    xc[idx] = (_Float16)v;
}

// ---------------------------------------------------------------------------
// Weight repacks (fp32 -> f16, zero-padded K)
// ---------------------------------------------------------------------------
__global__ void cast_f16_kernel(const float* __restrict__ s, _Float16* __restrict__ d, int n)
{
    int i = blockIdx.x * blockDim.x + threadIdx.x;
    if (i < n) d[i] = (_Float16)s[i];
}

// src: [Co][Cin][3] -> dst: [Co][Kp], k = dk*Cin+ci, zero for k >= 3*Cin
__global__ void repack_conv3_kernel(const float* __restrict__ src, _Float16* __restrict__ dst,
                                    int Cin, int Kp, int total)
{
    int idx = blockIdx.x * blockDim.x + threadIdx.x;
    if (idx >= total) return;
    int k  = idx % Kp;
    int co = idx / Kp;
    float v = 0.f;
    if (k < 3 * Cin) {
        int dk = k / Cin;
        int ci = k - dk * Cin;
        v = src[(co * Cin + ci) * 3 + dk];
    }
    dst[idx] = (_Float16)v;
}

// src: [Co][Cin] (k=1 conv) -> dst: [Co][Kp]
__global__ void repack_conv1_kernel(const float* __restrict__ src, _Float16* __restrict__ dst,
                                    int Cin, int Kp, int total)
{
    int idx = blockIdx.x * blockDim.x + threadIdx.x;
    if (idx >= total) return;
    int k  = idx % Kp;
    int co = idx / Kp;
    float v = (k < Cin) ? src[co * Cin + k] : 0.f;
    dst[idx] = (_Float16)v;
}

// fc_w [19][c*12+l] -> dst [32][l*128+c] (rows 19..31 zero)
__global__ void repack_fc_kernel(const float* __restrict__ src, _Float16* __restrict__ dst)
{
    int idx = blockIdx.x * blockDim.x + threadIdx.x;
    if (idx >= 32 * 1536) return;
    int k  = idx % 1536;
    int rr = idx / 1536;
    float v = 0.f;
    if (rr < 19) {
        int l = k / 128;
        int c = k % 128;
        v = src[rr * 1536 + c * 12 + l];
    }
    dst[idx] = (_Float16)v;
}

// ---------------------------------------------------------------------------
// Generic WMMA GEMM:  out[(b*Lo+lo)*Co + n] = sum_k A_row(b,lo)[k] * W[n][k]
// A_row(b,lo) = A + (b*SL + stride*lo + woff)*Cin   (contiguous window, Kp f16)
// W: [>=Co][Kp] f16 row-major (zero padded K).  4 waves/block, one 16x16 D tile
// per wave via v_wmma_f32_16x16x32_f16.
// ---------------------------------------------------------------------------
__global__ void __launch_bounds__(128)
gemm_wmma_kernel(const _Float16* __restrict__ A, const _Float16* __restrict__ W,
                 float* __restrict__ out,
                 int Lo, int Co, int Kp, int SL, int Cin, int stride, int woff)
{
    const int lane  = threadIdx.x & 31;
    const int wave  = threadIdx.x >> 5;
    const int rtile = blockIdx.x * 4 + wave;
    const int ctile = blockIdx.y;
    const int r  = lane & 15;
    const int hi = lane >> 4;          // half-wave select

    const int g  = rtile * 16 + r;     // global output row (M always mult of 64)
    const int b  = g / Lo;
    const int lo = g - b * Lo;
    const _Float16* arow = A + (size_t)(b * SL + stride * lo + woff) * Cin + hi * 16;
    const _Float16* wrow = W + (size_t)(ctile * 16 + r) * Kp + hi * 16;

    v8f acc = {};
    for (int k = 0; k < Kp; k += 32) {
        // A, contiguous k order used here (B matches): lane<16 holds K={k..k+15},
        //                        lane>=16 holds K={k+16..k+31}
        v8h alo = *(const v8h*)(arow + k);
        v8h ahi = *(const v8h*)(arow + k + 8);
        v16h av = __builtin_shufflevector(alo, ahi,
                                          0, 1, 2, 3, 4, 5, 6, 7,
                                          8, 9, 10, 11, 12, 13, 14, 15);
        // B 32x16 layout: lane n holds column n (row of W), K ascending
        v16h bv = *(const v16h*)(wrow + k);
        acc = __builtin_amdgcn_wmma_f32_16x16x32_f16(false, av, false, bv,
                                                     (short)0, acc, false, false);
    }

    // C/D layout: lane<16 -> M=v, N=lane ; lane>=16 -> M=8+v, N=lane-16
    const int n     = ctile * 16 + r;
    const int mbase = rtile * 16 + hi * 8;
#pragma unroll
    for (int v = 0; v < 8; ++v)
        out[(size_t)(mbase + v) * Co + n] = acc[v];
}

// ---------------------------------------------------------------------------
// BatchNorm statistics (training mode, biased variance) per channel.
// Optionally sums a second tensor (residual) on the fly.
// scale = g*rsqrt(var+eps); shift = beta - mean*scale
// ---------------------------------------------------------------------------
__global__ void bn_stats_kernel(const float* __restrict__ t, const float* __restrict__ sc,
                                const float* __restrict__ gam, const float* __restrict__ bet,
                                float* __restrict__ scale, float* __restrict__ shift,
                                int M, int C, int has_sc)
{
    __shared__ float sh1[256];
    __shared__ float sh2[256];
    const int c   = blockIdx.x;
    const int tid = threadIdx.x;
    float s = 0.f, s2 = 0.f;
    for (int i = tid; i < M; i += 256) {
        float v = t[(size_t)i * C + c];
        if (has_sc) v += sc[(size_t)i * C + c];
        s += v;
        s2 += v * v;
    }
    sh1[tid] = s;
    sh2[tid] = s2;
    __syncthreads();
    for (int o = 128; o > 0; o >>= 1) {
        if (tid < o) { sh1[tid] += sh1[tid + o]; sh2[tid] += sh2[tid + o]; }
        __syncthreads();
    }
    if (tid == 0) {
        float m   = sh1[0] / (float)M;
        float var = fmaxf(sh2[0] / (float)M - m * m, 0.f);
        float inv = rsqrtf(var + 1e-5f);
        float g   = gam[c];
        scale[c] = g * inv;
        shift[c] = bet[c] - m * g * inv;
    }
}

// ---------------------------------------------------------------------------
// Fused BN+ReLU (+optional residual add), writes zero-padded f16 activation
// [b][L+2][C] and optionally the fp32 unpadded [b][L][C] copy.
// ---------------------------------------------------------------------------
__global__ void bn_relu_pad_kernel(const float* __restrict__ t, const float* __restrict__ sc,
                                   const float* __restrict__ scale, const float* __restrict__ shift,
                                   _Float16* __restrict__ of16, float* __restrict__ of32,
                                   int Lo, int C, int total, int has_sc, int has_f32)
{
    int idx = blockIdx.x * blockDim.x + threadIdx.x;
    if (idx >= total) return;
    int c    = idx % C;
    int rest = idx / C;
    int l = rest % (Lo + 2);
    int b = rest / (Lo + 2);
    float r = 0.f;
    if (l >= 1 && l <= Lo) {
        int src = (b * Lo + (l - 1)) * C + c;
        float v = t[src];
        if (has_sc) v += sc[src];
        r = fmaxf(v * scale[c] + shift[c], 0.f);
        if (has_f32) of32[src] = r;
    }
    of16[idx] = (_Float16)r;
}

__global__ void fc_epilogue_kernel(const float* __restrict__ tfc, const float* __restrict__ fcb,
                                   float* __restrict__ out)
{
    int idx = blockIdx.x * blockDim.x + threadIdx.x;
    if (idx >= BATCH * 19) return;
    int rr = idx % 19;
    int b  = idx / 19;
    out[idx] = tfc[b * 32 + rr] + fcb[rr];
}

// ---------------------------------------------------------------------------
extern "C" void kernel_launch(void* const* d_in, const int* in_sizes, int n_in,
                              void* d_out, int out_size, void* d_ws, size_t ws_size,
                              hipStream_t stream)
{
    (void)in_sizes; (void)n_in; (void)out_size; (void)ws_size;

    // ---- input mapping (setup_inputs() insertion order) ----
    const int*   x    = (const int*)d_in[0];
    const int*   dd1  = (const int*)d_in[3];
    const int*   dd2  = (const int*)d_in[4];
    const float* emb  = (const float*)d_in[5];
    const float* pos  = (const float*)d_in[6];
    const float* w1   = (const float*)d_in[7];
    const float* bn1g = (const float*)d_in[9];
    const float* bn1b = (const float*)d_in[10];

    int bi[6][7];                 // c1_w, bn1_g, bn1_b, c2_w, bn2_g, bn2_b, sc_w
    int cins[6], couts[6], lins[6], los[6], downs[6];
    int pidx = 11, cin = 16, lin = 93;
    for (int k = 0; k < 6; ++k) {
        int down = (k % 2 == 0);
        int cout = down ? cin * 2 : cin;
        int lo   = down ? (lin + 1) / 2 : lin;
        bi[k][0] = pidx;
        bi[k][1] = pidx + 2;
        bi[k][2] = pidx + 3;
        bi[k][3] = pidx + 4;
        bi[k][4] = pidx + 6;
        bi[k][5] = pidx + 7;
        bi[k][6] = down ? pidx + 8 : -1;
        pidx += down ? 10 : 8;
        cins[k] = cin; couts[k] = cout; lins[k] = lin; los[k] = lo; downs[k] = down;
        cin = cout; lin = lo;
    }
    const float* fcw = (const float*)d_in[pidx];
    const float* fcb = (const float*)d_in[pidx + 1];

    // ---- workspace bump allocator (d_ws is 256B aligned) ----
    size_t off = 0;
    auto alloc = [&](size_t nbytes) -> void* {
        void* p = (char*)d_ws + off;
        off = (off + nbytes + 255) & ~(size_t)255;
        return p;
    };

    _Float16* xcf16 = (_Float16*)alloc((size_t)BATCH * SEQ * DIM * 2 + 512);
    _Float16* w1h   = (_Float16*)alloc((size_t)16 * 1600 * 2);
    _Float16* wfch  = (_Float16*)alloc((size_t)32 * 1536 * 2);

    _Float16 *wc1h[6], *wc2h[6], *wsch[6], *a1h[6], *yh[6];
    float* yf[6];
    int Kp1[6], Kp2[6], Kpsc[6];
    for (int k = 0; k < 6; ++k) {
        Kp1[k]  = (3 * cins[k] + 31) & ~31;
        Kp2[k]  = (3 * couts[k] + 31) & ~31;
        Kpsc[k] = (cins[k] + 31) & ~31;
        wc1h[k] = (_Float16*)alloc((size_t)couts[k] * Kp1[k] * 2);
        wc2h[k] = (_Float16*)alloc((size_t)couts[k] * Kp2[k] * 2);
        wsch[k] = downs[k] ? (_Float16*)alloc((size_t)couts[k] * Kpsc[k] * 2) : nullptr;
        size_t padded = (size_t)BATCH * (los[k] + 2) * couts[k];
        a1h[k] = (_Float16*)alloc(padded * 2 + 512);
        yh[k]  = (_Float16*)alloc(padded * 2 + 512);
        yf[k]  = (float*)alloc((size_t)BATCH * los[k] * couts[k] * 4);
    }
    _Float16* a0h = (_Float16*)alloc((size_t)BATCH * 95 * 16 * 2 + 512);
    float* t1  = (float*)alloc((size_t)196608 * 4);
    float* t2  = (float*)alloc((size_t)196608 * 4);
    float* tsc = (float*)alloc((size_t)196608 * 4);
    float* scaleb = (float*)alloc(128 * 4);
    float* shiftb = (float*)alloc(128 * 4);
    float* tfc    = (float*)alloc((size_t)BATCH * 32 * 4);

    auto cdiv = [](int a, int b) { return (a + b - 1) / b; };
    auto launch_gemm = [&](const _Float16* A, const _Float16* W, float* out,
                           int Lo, int Co, int Kp, int SL, int Cin_, int stride, int woff) {
        dim3 grid((BATCH * Lo) / 64, Co / 16);
        gemm_wmma_kernel<<<grid, dim3(128), 0, stream>>>(A, W, out, Lo, Co, Kp, SL, Cin_, stride, woff);
    };

    // ---- stage 0: gather + weight repacks ----
    {
        int total = BATCH * SEQ * DIM;
        gather_embed_kernel<<<cdiv(total, 256), 256, 0, stream>>>(x, dd1, dd2, emb, pos, xcf16, total);
        cast_f16_kernel<<<cdiv(16 * 1600, 256), 256, 0, stream>>>(w1, w1h, 16 * 1600);
        for (int k = 0; k < 6; ++k) {
            int t1n = couts[k] * Kp1[k];
            repack_conv3_kernel<<<cdiv(t1n, 256), 256, 0, stream>>>(
                (const float*)d_in[bi[k][0]], wc1h[k], cins[k], Kp1[k], t1n);
            int t2n = couts[k] * Kp2[k];
            repack_conv3_kernel<<<cdiv(t2n, 256), 256, 0, stream>>>(
                (const float*)d_in[bi[k][3]], wc2h[k], couts[k], Kp2[k], t2n);
            if (downs[k]) {
                int tsn = couts[k] * Kpsc[k];
                repack_conv1_kernel<<<cdiv(tsn, 256), 256, 0, stream>>>(
                    (const float*)d_in[bi[k][6]], wsch[k], cins[k], Kpsc[k], tsn);
            }
        }
        repack_fc_kernel<<<cdiv(32 * 1536, 256), 256, 0, stream>>>(fcw, wfch);
    }

    // ---- conv1 (16 x (4x400) GEMM) + BN1 + ReLU -> a0 ----
    launch_gemm(xcf16, w1h, t1, 93, 16, 1600, SEQ, DIM, 1, 0);
    bn_stats_kernel<<<16, 256, 0, stream>>>(t1, nullptr, bn1g, bn1b, scaleb, shiftb,
                                            BATCH * 93, 16, 0);
    {
        int total = BATCH * 95 * 16;
        bn_relu_pad_kernel<<<cdiv(total, 256), 256, 0, stream>>>(
            t1, nullptr, scaleb, shiftb, a0h, nullptr, 93, 16, total, 0, 0);
    }

    // ---- residual blocks ----
    const _Float16* curh = a0h;
    const float*    curf = nullptr;   // block 0 is a down block (fp32 input unused)
    for (int k = 0; k < 6; ++k) {
        int lo = los[k], co = couts[k], ci = cins[k], li = lins[k];
        int stride = downs[k] ? 2 : 1;
        int M = BATCH * lo;
        int total = BATCH * (lo + 2) * co;

        // conv1 (k=3, pad=1, stride s)
        launch_gemm(curh, wc1h[k], t1, lo, co, Kp1[k], li + 2, ci, stride, 0);
        bn_stats_kernel<<<co, 256, 0, stream>>>(t1, nullptr,
                                                (const float*)d_in[bi[k][1]],
                                                (const float*)d_in[bi[k][2]],
                                                scaleb, shiftb, M, co, 0);
        bn_relu_pad_kernel<<<cdiv(total, 256), 256, 0, stream>>>(
            t1, nullptr, scaleb, shiftb, a1h[k], nullptr, lo, co, total, 0, 0);

        // conv2 (k=3, pad=1, stride 1)
        launch_gemm(a1h[k], wc2h[k], t2, lo, co, Kp2[k], lo + 2, co, 1, 0);

        // shortcut
        const float* scp;
        if (downs[k]) {
            launch_gemm(curh, wsch[k], tsc, lo, co, Kpsc[k], li + 2, ci, 2, 1);
            scp = tsc;
        } else {
            scp = curf;   // identity: fp32 input of this block
        }

        // BN2 over (conv2 + shortcut), then ReLU
        bn_stats_kernel<<<co, 256, 0, stream>>>(t2, scp,
                                                (const float*)d_in[bi[k][4]],
                                                (const float*)d_in[bi[k][5]],
                                                scaleb, shiftb, M, co, 1);
        bn_relu_pad_kernel<<<cdiv(total, 256), 256, 0, stream>>>(
            t2, scp, scaleb, shiftb, yh[k], yf[k], lo, co, total, 1, 1);

        curh = yh[k];
        curf = yf[k];
    }

    // ---- FC head: [128 x 1536] @ [1536 x 19] via WMMA (N padded to 32) ----
    launch_gemm(curh, wfch, tfc, /*Lo=*/1, /*Co=*/32, /*Kp=*/1536,
                /*SL=*/14, /*Cin=*/128, /*stride=*/1, /*woff=*/1);
    fc_epilogue_kernel<<<cdiv(BATCH * 19, 256), 256, 0, stream>>>(tfc, fcb, (float*)d_out);
}
